// JointLoss_5085241278662
// MI455X (gfx1250) — matrix operand order repaired and checked by reading, not compiled, not run on hardware
//
#include <hip/hip_runtime.h>
#include <hip/hip_bf16.h>

// ---------------------------------------------------------------------------
// JointLoss (NT-Xent + two MSEs) for MI455X / gfx1250, wave32.
//   sim = z z^T via v_wmma_f32_16x16x32_bf16.
//   B panels staged to LDS with global_load_async_to_lds_b128 (double buffer,
//   ASYNCcnt + s_wait_asynccnt + barrier), shared by 8 waves per workgroup.
//   |sim|<=1 => fixed logsumexp shift of 2.0 (no online max).
// ---------------------------------------------------------------------------

typedef __attribute__((ext_vector_type(16))) __bf16       v16bf;
typedef __attribute__((ext_vector_type(8)))  float        v8f;
typedef __attribute__((ext_vector_type(4)))  unsigned int v4u;

#define NROWS 4096
#define HALFN 2048
#define DIM   512
#define NT    256        // 16-wide tiles along rows/cols
#define NSPLIT 8         // grid.y: N-tiles split per block
#define NT_PER (NT / NSPLIT)

union AFrag { v16bf v; v4u u[2]; };

// ws layout (floats): [0..3]=accs(closs,recon,zrecon), [64..4159]=rowacc,
// [4160..8255]=rowpos, then z_bf16 at byte offset 33024.
#define WS_ZERO_FLOATS (64 + 2 * NROWS)
#define WS_Z_BYTE_OFF  (256 + 2 * NROWS * 4)

// ---- zero the accumulators / per-row partials ------------------------------
__global__ void jl_init(float* w) {
  const int i = blockIdx.x * blockDim.x + threadIdx.x;
  if (i < WS_ZERO_FLOATS) w[i] = 0.f;
}

// ---- row L2-normalize + cast to bf16 (one wave per row) --------------------
__global__ __launch_bounds__(256) void jl_norm_cast(const float* __restrict__ rep,
                                                    __bf16* __restrict__ z) {
  const int wave = threadIdx.x >> 5;
  const int lane = threadIdx.x & 31;
  const int row  = blockIdx.x * 8 + wave;
  const float4* x4 = (const float4*)(rep + (size_t)row * DIM);

  float4 d[4];
  float  s = 0.f;
#pragma unroll
  for (int t = 0; t < 4; ++t) {
    d[t] = x4[lane + t * 32];
    s += d[t].x * d[t].x + d[t].y * d[t].y + d[t].z * d[t].z + d[t].w * d[t].w;
  }
#pragma unroll
  for (int m = 1; m < 32; m <<= 1) s += __shfl_xor(s, m, 32);

  const float inv = 1.f / fmaxf(sqrtf(s), 1e-8f);
  __bf16* zr = z + (size_t)row * DIM;
#pragma unroll
  for (int t = 0; t < 4; ++t) {
    const int base = (lane + t * 32) * 4;
    zr[base + 0] = (__bf16)(d[t].x * inv);
    zr[base + 1] = (__bf16)(d[t].y * inv);
    zr[base + 2] = (__bf16)(d[t].z * inv);
    zr[base + 3] = (__bf16)(d[t].w * inv);
  }
}

// ---- grid-stride sum of squared differences (float4) -----------------------
__global__ __launch_bounds__(256) void jl_mse(const float* __restrict__ a,
                                              const float* __restrict__ b,
                                              int n4, float* __restrict__ acc) {
  const int idx    = blockIdx.x * blockDim.x + threadIdx.x;
  const int stride = gridDim.x * blockDim.x;
  float s = 0.f;
  for (int i = idx; i < n4; i += stride) {
    const float4 av = ((const float4*)a)[i];
    const float4 bv = ((const float4*)b)[i];
    const float dx = av.x - bv.x, dy = av.y - bv.y;
    const float dz = av.z - bv.z, dw = av.w - bv.w;
    s += dx * dx + dy * dy + dz * dz + dw * dw;
  }
#pragma unroll
  for (int m = 1; m < 32; m <<= 1) s += __shfl_xor(s, m, 32);
  if ((threadIdx.x & 31) == 0) atomicAdd(acc, s);
}

// ---- async-stage one 16-col x 512-K bf16 panel (16KB) into LDS -------------
// 256 threads x 4 x 16B chunks. LDS layout: col*1024 + kByte (two 16B pieces
// 32B apart per lane-fragment). Tracked by ASYNCcnt of the issuing wave.
__device__ __forceinline__ void jl_stage_panel(unsigned long long zbase, int nt,
                                               unsigned lds_base) {
  const int t = threadIdx.x;  // 0..255
#pragma unroll
  for (int q = 0; q < 4; ++q) {
    const int ch  = t + q * 256;          // chunk 0..1023
    const int col = ch >> 6;              // 0..15
    const int off = (ch & 63) * 16;       // byte within the 1KB column
    const unsigned voff  = (unsigned)((nt * 16 + col) * (DIM * 2) + off);
    const unsigned laddr = lds_base + (unsigned)(col * 1024 + off);
    asm volatile("global_load_async_to_lds_b128 %0, %1, %2 offset:0"
                 :: "v"(laddr), "v"(voff), "s"(zbase)
                 : "memory");
  }
}

// ---- contrastive: WMMA sim tiles, LDS-shared B, fused exp-sum --------------
// Block: 8 waves, wave w owns M-tile blockIdx.x*8+w. grid.y splits N range.
__global__ __launch_bounds__(256) void jl_xneg(const __bf16* __restrict__ z,
                                               float* __restrict__ rowacc,
                                               float* __restrict__ rowpos) {
  __shared__ v4u smem[2][1024];  // 2 x 16KB panels

  const int lane = threadIdx.x & 31;
  const int wave = threadIdx.x >> 5;
  const int r16  = lane & 15;
  const int hi   = lane >> 4;
  const int mt   = blockIdx.x * 8 + wave;

  // A fragments: lane holds row (mt*16 + r16); per 32-K chunk two 16B pieces
  // at +hi*8 and +16+hi*8 halfwords (ISA 16-bit A layout). Kept resident.
  const __bf16* arow = z + (size_t)(mt * 16 + r16) * DIM;
  AFrag a[16];
#pragma unroll
  for (int k = 0; k < 16; ++k) {
    const int kb = k * 32 + hi * 8;
    a[k].u[0] = *(const v4u*)(arow + kb);
    a[k].u[1] = *(const v4u*)(arow + kb + 16);
  }

  float accv[8], posv[8];
#pragma unroll
  for (int v = 0; v < 8; ++v) { accv[v] = 0.f; posv[v] = 0.f; }

  const int myrow = mt * 16 + hi * 8;          // C rows for this lane half
  const unsigned lds0 = (unsigned)(uintptr_t)(&smem[0][0]);
  const unsigned lds1 = (unsigned)(uintptr_t)(&smem[1][0]);
  const unsigned long long zb = (unsigned long long)(uintptr_t)z;
  const int nt0 = blockIdx.y * NT_PER;

  jl_stage_panel(zb, nt0, lds0);
  asm volatile("s_wait_asynccnt 0" ::: "memory");
  __syncthreads();

  int buf = 0;
  for (int it = 0; it < NT_PER; ++it) {
    const int nt = nt0 + it;
    if (it + 1 < NT_PER) jl_stage_panel(zb, nt + 1, buf ? lds0 : lds1);

    // B fragment from LDS: col r16, 16B pieces at +hi*8 and +16+hi*8 halfwords
    const v4u* bpan = &smem[buf][0];
    const int  bb   = r16 * 64 + hi;
    v8f c = {};
#pragma unroll
    for (int k = 0; k < 16; ++k) {
      AFrag b;
      b.u[0] = bpan[bb + k * 4];
      b.u[1] = bpan[bb + k * 4 + 2];
      c = __builtin_amdgcn_wmma_f32_16x16x32_bf16(
          false, a[k].v, false, b.v, (short)0, c, false, false);
    }
    // epilogue: c[v] = sim[myrow+v, nt*16 + r16]
#pragma unroll
    for (int v = 0; v < 8; ++v) {
      const int i = myrow + v;
      const int j = nt * 16 + r16;
      const float s = c[v];
      if (j != i)           accv[v] += __expf(2.f * s - 2.f);  // s/tau - 2
      if (j == (i ^ HALFN)) posv[v]  = s;                      // positive pair
    }

    asm volatile("s_wait_asynccnt 0" ::: "memory");  // next panel landed
    __syncthreads();                                 // everyone done with buf
    buf ^= 1;
  }

  // reduce the 16 column-classes per row (stays within each 16-lane half)
#pragma unroll
  for (int v = 0; v < 8; ++v) {
#pragma unroll
    for (int m = 1; m < 16; m <<= 1) {
      accv[v] += __shfl_xor(accv[v], m, 32);
      posv[v] += __shfl_xor(posv[v], m, 32);
    }
  }
  if (r16 == 0) {
#pragma unroll
    for (int v = 0; v < 8; ++v) {
      atomicAdd(rowacc + myrow + v, accv[v]);
      atomicAdd(rowpos + myrow + v, posv[v]);
    }
  }
}

// ---- per-row loss: 2 + log(sum_exp) - 2*pos, reduce to closs sum -----------
__global__ __launch_bounds__(256) void jl_rowloss(const float* __restrict__ rowacc,
                                                  const float* __restrict__ rowpos,
                                                  float* __restrict__ acc) {
  const int i = blockIdx.x * 256 + threadIdx.x;  // 0..4095
  float l = (2.f + __logf(rowacc[i])) - 2.f * rowpos[i];
#pragma unroll
  for (int m = 1; m < 32; m <<= 1) l += __shfl_xor(l, m, 32);
  if ((threadIdx.x & 31) == 0) atomicAdd(acc, l);
}

// ---- combine ---------------------------------------------------------------
__global__ void jl_finalize(const float* __restrict__ acc, float* __restrict__ out) {
  if (threadIdx.x == 0) {
    const float closs  = acc[0] / (float)NROWS;   // /2N
    const float recon  = acc[1] / (float)NROWS;   // /target.shape[0]
    const float zrecon = acc[2] / (float)HALFN;   // /N
    out[0] = closs + recon + zrecon;
    out[1] = closs;
    out[2] = recon;
    out[3] = zrecon;
  }
}

extern "C" void kernel_launch(void* const* d_in, const int* in_sizes, int n_in,
                              void* d_out, int out_size, void* d_ws, size_t ws_size,
                              hipStream_t stream) {
  const float* rep    = (const float*)d_in[0];  // (4096, 512)
  const float* xrecon = (const float*)d_in[1];  // (4096, 1024)
  const float* xorig  = (const float*)d_in[2];  // (4096, 1024)
  float* out = (float*)d_out;

  float*  wsf    = (float*)d_ws;
  float*  accs   = wsf;                 // 4 floats
  float*  rowacc = wsf + 64;            // 4096 floats
  float*  rowpos = wsf + 64 + NROWS;    // 4096 floats
  __bf16* z      = (__bf16*)((char*)d_ws + WS_Z_BYTE_OFF);  // 4 MB

  jl_init<<<(WS_ZERO_FLOATS + 255) / 256, 256, 0, stream>>>(wsf);
  jl_norm_cast<<<NROWS / 8, 256, 0, stream>>>(rep, z);
  jl_mse<<<1024, 256, 0, stream>>>(xrecon, xorig, (NROWS * 1024) / 4, accs + 1);
  jl_mse<<<256, 256, 0, stream>>>(rep, rep + (size_t)HALFN * DIM,
                                  (HALFN * DIM) / 4, accs + 2);
  jl_xneg<<<dim3(NROWS / 128, NSPLIT), 256, 0, stream>>>(z, rowacc, rowpos);
  jl_rowloss<<<NROWS / 256, 256, 0, stream>>>(rowacc, rowpos, accs);
  jl_finalize<<<1, 32, 0, stream>>>(accs, out);
}
